// SpikingMNIST_48885317763693
// MI455X (gfx1250) — compile-verified, matrix operand order checked by reading, not run on hardware
//
#include <hip/hip_runtime.h>

typedef _Float16 v16h __attribute__((ext_vector_type(16)));
typedef _Float16 v8h  __attribute__((ext_vector_type(8)));
typedef _Float16 v4h  __attribute__((ext_vector_type(4)));
typedef float    v8f  __attribute__((ext_vector_type(8)));

#define SNN_BETA 0.95f
#define SNN_THR  1.0f
#define TSTEPS   25
#define K1       784
#define HID      128
#define OUTC     10
#define WG_M     128   // batch rows per workgroup (8 waves x 16)
#define KC       32    // K chunk per f16 WMMA
#define NCHUNKS  25    // ceil(784/32); last chunk half, zero-padded
#define XS       40    // padded LDS stride (halves) for x/W1 tiles: 80B, 16B-aligned, conflict-free
#define SS       136   // padded LDS stride (halves) for spk/W2: 272B, 16B-aligned, conflict-free

// Load a 16x32 f16 A/B fragment (v16h) from LDS: two ds_load_b128 per lane.
__device__ __forceinline__ v16h ld_frag(const _Float16* p) {
  v8h lo = *reinterpret_cast<const v8h*>(p);
  v8h hi = *reinterpret_cast<const v8h*>(p + 16);
  v16h r;
#pragma unroll
  for (int i = 0; i < 8; ++i) { r[i] = lo[i]; r[i + 8] = hi[i]; }
  return r;
}

__global__ __launch_bounds__(256, 1) void snn_fused_kernel(
    const float* __restrict__ x, const float* __restrict__ w1,
    const float* __restrict__ w2, float* __restrict__ out, int Btot)
{
  __shared__ __align__(16) _Float16 sX [WG_M * XS];   // x tile   (f16)
  __shared__ __align__(16) _Float16 sW1[HID  * XS];   // W1 tile  (f16)
  __shared__ __align__(16) _Float16 sSpk[WG_M * SS];  // spike tile (f16)
  __shared__ __align__(16) _Float16 sW2[16 * SS];     // W2 padded 16x128 (f16)

  const int tid   = threadIdx.x;
  const int lane  = tid & 31;
  const int wave  = tid >> 5;
  const int lrow  = lane & 15;            // M (A) / N (B,C,D) index within tile
  const int khalf = (lane >> 4) * 8;      // K sub-group select for A/B fragments
  const int mhi   = (lane >> 4) * 8;      // C/D: lanes 16-31 hold M = r+8
  const int wgM   = blockIdx.x * WG_M;

  // ---- stage W2 as padded 16x128 f16 (rows 10..15 zero) ----
#pragma unroll
  for (int i = 0; i < 8; ++i) {
    int lin = tid + i * 256;              // 0..2047
    int row = lin >> 7;
    int col = lin & 127;
    float v = (row < OUTC) ? w2[row * HID + col] : 0.0f;
    sW2[row * SS + col] = (_Float16)v;
  }

  // ---- phase 1: cur1 = x @ W1^T via v_wmma_f32_16x16x32_f16 ----
  // Software-pipelined: global loads for chunk c+1 fly while WMMAs chew chunk c.
  const v8f vzero = {0.f, 0.f, 0.f, 0.f, 0.f, 0.f, 0.f, 0.f};
  v8f acc[8];
#pragma unroll
  for (int nt = 0; nt < 8; ++nt) acc[nt] = vzero;

  float4 rx[4], rw[4];                    // in-flight staging registers

  auto load_chunk = [&](int c) {
    const int k0 = c * KC;
#pragma unroll
    for (int i = 0; i < 4; ++i) {
      int lin = tid + i * 256;            // 0..1023
      int row = lin >> 3;                 // 0..127
      int cg  = (lin & 7) << 2;           // 0,4,...,28
      int k   = k0 + cg;
      rx[i] = {0.f, 0.f, 0.f, 0.f};
      rw[i] = {0.f, 0.f, 0.f, 0.f};
      if (k < K1) {                       // zero-pad tail of last chunk
        rx[i] = *reinterpret_cast<const float4*>(x  + (size_t)(wgM + row) * K1 + k);
        rw[i] = *reinterpret_cast<const float4*>(w1 + (size_t)row * K1 + k);
        if (i == 0 && k + 2 * KC < K1)    // warm L2 two chunks ahead of this load
          __builtin_prefetch(x + (size_t)(wgM + row) * K1 + k + 2 * KC, 0, 1);
      }
    }
  };

  auto store_chunk = [&]() {              // fp32 regs -> f16 LDS tiles
#pragma unroll
    for (int i = 0; i < 4; ++i) {
      int lin = tid + i * 256;
      int row = lin >> 3;
      int cg  = (lin & 7) << 2;
      v4h hx, hw;
      hx[0] = (_Float16)rx[i].x; hx[1] = (_Float16)rx[i].y;
      hx[2] = (_Float16)rx[i].z; hx[3] = (_Float16)rx[i].w;
      hw[0] = (_Float16)rw[i].x; hw[1] = (_Float16)rw[i].y;
      hw[2] = (_Float16)rw[i].z; hw[3] = (_Float16)rw[i].w;
      *reinterpret_cast<v4h*>(sX  + row * XS + cg) = hx;
      *reinterpret_cast<v4h*>(sW1 + row * XS + cg) = hw;
    }
  };

  load_chunk(0);                          // prologue

#pragma unroll 1
  for (int c = 0; c < NCHUNKS; ++c) {
    store_chunk();                        // waits on loads issued last iteration
    __syncthreads();

    if (c + 1 < NCHUNKS) load_chunk(c + 1);  // issue next chunk's global loads now

    v16h afrag = ld_frag(sX + (wave * 16 + lrow) * XS + khalf);
#pragma unroll
    for (int nt = 0; nt < 8; ++nt) {
      v16h bfrag = ld_frag(sW1 + (nt * 16 + lrow) * XS + khalf);
      acc[nt] = __builtin_amdgcn_wmma_f32_16x16x32_f16(
          false, afrag, false, bfrag, (short)0, acc[nt], false, false);
    }
    __syncthreads();                      // all waves done reading before overwrite
  }

  // ---- preload W2^T B-fragments (K=128 -> 4 chunks), step-invariant ----
  v16h b2[4];
#pragma unroll
  for (int kc = 0; kc < 4; ++kc)
    b2[kc] = ld_frag(sW2 + lrow * SS + kc * 32 + khalf);

  // ---- phase 2: 25 LIF timesteps fully on-chip ----
  float mem1[8][8];
#pragma unroll
  for (int nt = 0; nt < 8; ++nt)
#pragma unroll
    for (int r = 0; r < 8; ++r) mem1[nt][r] = 0.0f;
  float mem2[8];
#pragma unroll
  for (int r = 0; r < 8; ++r) mem2[r] = 0.0f;

  const int o = lrow;                      // output column for this lane (C/D layout)

#pragma unroll 1
  for (int t = 0; t < TSTEPS; ++t) {
    // LIF layer 1; reset = (mem_old > thr) == previous spike (strict >, mem0 = 0)
#pragma unroll
    for (int nt = 0; nt < 8; ++nt) {
      const int h = nt * 16 + lrow;
#pragma unroll
      for (int r = 0; r < 8; ++r) {
        float m = mem1[nt][r];
        float reset = (m > SNN_THR) ? 1.0f : 0.0f;
        m = SNN_BETA * m + acc[nt][r] - reset;
        mem1[nt][r] = m;
        // wave-local LDS region: rows [wave*16, wave*16+16) only -> no barrier
        sSpk[(wave * 16 + mhi + r) * SS + h] =
            (m > SNN_THR) ? (_Float16)1.0f : (_Float16)0.0f;
      }
    }

    // cur2 = spk1 @ W2^T : one 16x16 tile per wave, K=128 = 4 WMMAs
    v8f c2 = vzero;
#pragma unroll
    for (int kc = 0; kc < 4; ++kc) {
      v16h a2 = ld_frag(sSpk + (wave * 16 + lrow) * SS + kc * 32 + khalf);
      c2 = __builtin_amdgcn_wmma_f32_16x16x32_f16(
          false, a2, false, b2[kc], (short)0, c2, false, false);
    }

    // LIF layer 2 + emit spk2 to HBM (only valid output columns o < 10)
    float* outt = out + (size_t)t * Btot * OUTC;
#pragma unroll
    for (int r = 0; r < 8; ++r) {
      float m = mem2[r];
      float reset = (m > SNN_THR) ? 1.0f : 0.0f;
      m = SNN_BETA * m + c2[r] - reset;
      mem2[r] = m;
      if (o < OUTC) {
        int b = wgM + wave * 16 + mhi + r;
        outt[(size_t)b * OUTC + o] = (m > SNN_THR) ? 1.0f : 0.0f;
      }
    }
  }
}

extern "C" void kernel_launch(void* const* d_in, const int* in_sizes, int n_in,
                              void* d_out, int out_size, void* d_ws, size_t ws_size,
                              hipStream_t stream) {
  const float* x  = (const float*)d_in[0];   // [B, 784] f32
  const float* w1 = (const float*)d_in[1];   // [128, 784] f32
  const float* w2 = (const float*)d_in[2];   // [10, 128] f32
  float* out = (float*)d_out;                // [25, B, 10] f32
  const int Btot = in_sizes[0] / K1;         // 32768
  const int grid = Btot / WG_M;              // 256 workgroups
  snn_fused_kernel<<<grid, 256, 0, stream>>>(x, w1, w2, out, Btot);
}
